// RNNDecoderAttention_10514079941510
// MI455X (gfx1250) — compile-verified, matrix operand order checked by reading, not run on hardware
//
#include <hip/hip_runtime.h>
#include <hip/hip_bf16.h>

// ---------------------------------------------------------------------------
// RNN decoder step + Bahdanau attention for MI455X (gfx1250).
// B=256, L=512, D_IN=512, H=512, OUT=4096.
// GEMMs via v_wmma_f32_16x16x32_bf16 (fp32 accumulate); h_enc streaming
// kernels are bandwidth-bound float4 paths (2 x 537MB HBM passes ~= 46us).
// ---------------------------------------------------------------------------

typedef __attribute__((ext_vector_type(16))) __bf16 v16bf;
typedef __attribute__((ext_vector_type(8)))  float  v8f;
typedef __attribute__((ext_vector_type(4)))  int    i32x4;

#define Bn    256
#define Ln    512
#define Hn    512
#define OUTn  4096

// ------------------------- fp32 -> bf16 conversion -------------------------
__global__ void cvt_f32_bf16(const float* __restrict__ src,
                             __bf16* __restrict__ dst, int n) {
  int i = blockIdx.x * blockDim.x + threadIdx.x;
  if (i < n) dst[i] = (__bf16)src[i];
}

// K-concat two row-major fp32 matrices into one bf16 matrix (ld = ka+kb).
__global__ void concat_rows_bf16(const float* __restrict__ a,
                                 const float* __restrict__ b,
                                 __bf16* __restrict__ dst,
                                 int ka, int kb, int total) {
  int i = blockIdx.x * blockDim.x + threadIdx.x;
  if (i >= total) return;
  int ld  = ka + kb;
  int row = i / ld;
  int col = i - row * ld;
  float v = (col < ka) ? a[(size_t)row * ka + col]
                       : b[(size_t)row * kb + (col - ka)];
  dst[i] = (__bf16)v;
}

// ------------------------- bf16 WMMA GEMM ----------------------------------
// C[M,N] = A[M,K] * Bw[N,K]^T (+bias0 +bias1), all row-major, K contiguous.
// A-fragment layout (16x32 bf16): lane m%16 holds row m; lanes 0-15 carry
// K = {k0..k0+7, k0+16..k0+23}, lanes 16-31 carry K = {+8..+15, +24..+31}.
// B-fragment is symmetric (lane n holds column n of B == row n of Bw).
__device__ __forceinline__ v16bf load_frag(const __bf16* __restrict__ p,
                                           int ld, int row0, int k0) {
  int lane = threadIdx.x & 31;
  int r    = row0 + (lane & 15);
  int kb   = k0 + ((lane >> 4) << 3);
  const __bf16* q = p + (size_t)r * ld + kb;
  v16bf v;
  i32x4* pv = reinterpret_cast<i32x4*>(&v);
  pv[0] = *reinterpret_cast<const i32x4*>(q);        // K  kb .. kb+7
  pv[1] = *reinterpret_cast<const i32x4*>(q + 16);   // K  kb+16 .. kb+23
  return v;
}

// One wave (32 threads) per 16x64 output tile; A-fragment reused 4x.
__global__ void wmma_gemm_bf16(const __bf16* __restrict__ A, int lda,
                               const __bf16* __restrict__ Bw, int ldb,
                               const float* __restrict__ bias0,
                               const float* __restrict__ bias1,
                               float* __restrict__ C, int ldc, int K) {
  const int row0 = blockIdx.y * 16;
  const int col0 = blockIdx.x * 64;
  v8f acc0 = {}, acc1 = {}, acc2 = {}, acc3 = {};

  for (int k0 = 0; k0 < K; k0 += 32) {
    v16bf a  = load_frag(A, lda, row0, k0);
    v16bf b0 = load_frag(Bw, ldb, col0 +  0, k0);
    v16bf b1 = load_frag(Bw, ldb, col0 + 16, k0);
    v16bf b2 = load_frag(Bw, ldb, col0 + 32, k0);
    v16bf b3 = load_frag(Bw, ldb, col0 + 48, k0);
    acc0 = __builtin_amdgcn_wmma_f32_16x16x32_bf16(false, a, false, b0, (short)0, acc0, false, false);
    acc1 = __builtin_amdgcn_wmma_f32_16x16x32_bf16(false, a, false, b1, (short)0, acc1, false, false);
    acc2 = __builtin_amdgcn_wmma_f32_16x16x32_bf16(false, a, false, b2, (short)0, acc2, false, false);
    acc3 = __builtin_amdgcn_wmma_f32_16x16x32_bf16(false, a, false, b3, (short)0, acc3, false, false);
  }

  // C/D layout: VGPR r, lanes 0-15 -> M=r, N=lane; lanes 16-31 -> M=r+8.
  const int lane = threadIdx.x & 31;
  const int ncol = lane & 15;
  const int mrow = row0 + ((lane >> 4) << 3);
  v8f accs[4] = {acc0, acc1, acc2, acc3};
#pragma unroll
  for (int i = 0; i < 4; ++i) {
    int n = col0 + 16 * i + ncol;
    float bsum = (bias0 ? bias0[n] : 0.0f) + (bias1 ? bias1[n] : 0.0f);
#pragma unroll
    for (int r = 0; r < 8; ++r)
      C[(size_t)(mrow + r) * ldc + n] = accs[i][r] + bsum;
  }
}

// ------------------------- LSTM pointwise gates ----------------------------
__device__ __forceinline__ float sigm(float x) { return 1.0f / (1.0f + __expf(-x)); }

__global__ void lstm_pointwise(const float* __restrict__ z,
                               const float* __restrict__ c0,
                               float* __restrict__ h_out,
                               float* __restrict__ c_out,
                               __bf16* __restrict__ h_bf,   // (B,H) ld=H
                               __bf16* __restrict__ A2) {   // (B,1536); h at col 1024
  int idx = blockIdx.x * blockDim.x + threadIdx.x;   // B*H threads
  int b = idx >> 9, n = idx & 511;
  const float* zr = z + (size_t)b * (4 * Hn);
  float ig = sigm(zr[n]);
  float fg = sigm(zr[Hn + n]);
  float gg = tanhf(zr[2 * Hn + n]);
  float og = sigm(zr[3 * Hn + n]);
  float c  = fg * c0[idx] + ig * gg;
  float h  = og * tanhf(c);
  c_out[idx] = c;
  h_out[idx] = h;
  h_bf[idx]  = (__bf16)h;
  A2[(size_t)b * (3 * Hn) + 2 * Hn + n] = (__bf16)h;
}

// -------------------- attention scores: e[b,l] = Vh[b].h_enc[l,b,:] --------
__global__ void attn_scores(const float* __restrict__ Vh,
                            const float* __restrict__ henc,
                            float* __restrict__ e) {
  int l = blockIdx.x, b = blockIdx.y, lane = threadIdx.x;  // 1 wave / (b,l)
  const float4* p = reinterpret_cast<const float4*>(henc + ((size_t)l * Bn + b) * (2 * Hn));
  const float4* v = reinterpret_cast<const float4*>(Vh + (size_t)b * (2 * Hn));
  float s = 0.0f;
#pragma unroll
  for (int j = 0; j < 8; ++j) {                 // 2H=1024 floats = 256 float4
    float4 x = p[lane + 32 * j];
    float4 y = v[lane + 32 * j];
    s += x.x * y.x + x.y * y.y + x.z * y.z + x.w * y.w;
  }
#pragma unroll
  for (int off = 16; off; off >>= 1) s += __shfl_xor(s, off, 32);
  if (lane == 0) e[(size_t)b * Ln + l] = s;
}

// -------------------- softmax over L per batch row -------------------------
__global__ void softmax_rows(const float* __restrict__ e,
                             float* __restrict__ alpha) {
  __shared__ float red[256];
  int b = blockIdx.x, t = threadIdx.x;           // 256 threads, 2 elems each
  const float* er = e + (size_t)b * Ln;
  float v0 = er[t], v1 = er[t + 256];
  red[t] = fmaxf(v0, v1);
  __syncthreads();
  for (int s = 128; s; s >>= 1) { if (t < s) red[t] = fmaxf(red[t], red[t + s]); __syncthreads(); }
  float m = red[0];
  __syncthreads();
  float e0 = __expf(v0 - m), e1 = __expf(v1 - m);
  red[t] = e0 + e1;
  __syncthreads();
  for (int s = 128; s; s >>= 1) { if (t < s) red[t] += red[t + s]; __syncthreads(); }
  float inv = 1.0f / red[0];
  alpha[(size_t)b * Ln + t]       = e0 * inv;
  alpha[(size_t)b * Ln + t + 256] = e1 * inv;
}

// ------------- context: c_vec[b,:] = sum_l alpha[b,l]*view[b,l,:] ----------
// view is the raw reinterpret: flat[b*L*2H + l*2H + d]
__global__ void attn_context(const float* __restrict__ alpha,
                             const float* __restrict__ henc,
                             float* __restrict__ cvec,
                             __bf16* __restrict__ A2) {    // (B,1536); cvec at col 0
  int b = blockIdx.x, t = threadIdx.x;           // 256 threads * float4 = 1024
  const float* base = henc + (size_t)b * Ln * (2 * Hn);
  const float* ar   = alpha + (size_t)b * Ln;
  float4 acc = {0.f, 0.f, 0.f, 0.f};
  for (int l = 0; l < Ln; ++l) {
    if (l + 8 < Ln)
      __builtin_prefetch(base + (size_t)(l + 8) * (2 * Hn) + t * 4, 0, 0);
    float a = ar[l];
    float4 x = reinterpret_cast<const float4*>(base + (size_t)l * (2 * Hn))[t];
    acc.x += a * x.x; acc.y += a * x.y; acc.z += a * x.z; acc.w += a * x.w;
  }
  reinterpret_cast<float4*>(cvec + (size_t)b * (2 * Hn))[t] = acc;
  __bf16* a2 = A2 + (size_t)b * (3 * Hn) + t * 4;
  a2[0] = (__bf16)acc.x; a2[1] = (__bf16)acc.y;
  a2[2] = (__bf16)acc.z; a2[3] = (__bf16)acc.w;
}

// ---------------------------------------------------------------------------
extern "C" void kernel_launch(void* const* d_in, const int* in_sizes, int n_in,
                              void* d_out, int out_size, void* d_ws, size_t ws_size,
                              hipStream_t stream) {
  const float* x     = (const float*)d_in[0];   // (1,B,D_IN)
  const float* h0    = (const float*)d_in[1];   // (1,B,H)
  const float* c0    = (const float*)d_in[2];   // (1,B,H)
  const float* henc  = (const float*)d_in[3];   // (L,B,2H)
  const float* W_ih  = (const float*)d_in[4];   // (4H,D_IN)
  const float* W_hh  = (const float*)d_in[5];   // (4H,H)
  const float* b_ih  = (const float*)d_in[6];   // (4H)
  const float* b_hh  = (const float*)d_in[7];   // (4H)
  const float* V_w   = (const float*)d_in[8];   // (2H,H)
  const float* V_b   = (const float*)d_in[9];   // (2H)
  const float* fc_w  = (const float*)d_in[10];  // (OUT,3H)
  const float* fc_b  = (const float*)d_in[11];  // (OUT)

  float* out_main = (float*)d_out;                     // (B,OUT)
  float* out_h    = out_main + (size_t)Bn * OUTn;      // (1,B,H)
  float* out_c    = out_h + (size_t)Bn * Hn;           // (1,B,H)

  // ---- workspace layout (bytes) ----
  char* w = (char*)d_ws;
  __bf16* A1    = (__bf16*)(w + 0x000000);  // (B,1024)  [x|h0]      512 KB
  __bf16* Wcatb = (__bf16*)(w + 0x080000);  // (4H,1024) [Wih|Whh]     4 MB
  float*  zbuf  = (float* )(w + 0x480000);  // (B,4H)                  2 MB
  __bf16* Ahb   = (__bf16*)(w + 0x680000);  // (B,H) h_new bf16      256 KB
  __bf16* Vwb   = (__bf16*)(w + 0x6C0000);  // (2H,H)                  1 MB
  float*  Vh    = (float* )(w + 0x7C0000);  // (B,2H)                  1 MB
  float*  ebuf  = (float* )(w + 0x8C0000);  // (B,L)                 512 KB
  float*  albuf = (float* )(w + 0x940000);  // (B,L)                 512 KB
  float*  cvec  = (float* )(w + 0x9C0000);  // (B,2H)                  1 MB
  __bf16* A2    = (__bf16*)(w + 0xAC0000);  // (B,3H) [cvec|h_new]   768 KB
  __bf16* fcwb  = (__bf16*)(w + 0xB80000);  // (OUT,3H)               12 MB

  // ---- 1. build bf16 operands ----
  {
    int tot = Bn * 1024;                               // [x | h0]
    concat_rows_bf16<<<(tot + 255) / 256, 256, 0, stream>>>(x, h0, A1, 512, 512, tot);
  }
  {
    int tot = 4 * Hn * 1024;                           // [W_ih | W_hh]
    concat_rows_bf16<<<(tot + 255) / 256, 256, 0, stream>>>(W_ih, W_hh, Wcatb, 512, 512, tot);
  }
  {
    int n = 2 * Hn * Hn;                               // V_w
    cvt_f32_bf16<<<(n + 255) / 256, 256, 0, stream>>>(V_w, Vwb, n);
  }
  {
    int n = OUTn * 3 * Hn;                             // fc_w
    cvt_f32_bf16<<<(n + 255) / 256, 256, 0, stream>>>(fc_w, fcwb, n);
  }

  // ---- 2. LSTM gates GEMM: z = [x|h] @ [Wih|Whh]^T + b_ih + b_hh ----
  wmma_gemm_bf16<<<dim3((4 * Hn) / 64, Bn / 16), 32, 0, stream>>>(
      A1, 1024, Wcatb, 1024, b_ih, b_hh, zbuf, 4 * Hn, 1024);

  // ---- 3. gates / cell update ----
  lstm_pointwise<<<(Bn * Hn) / 256, 256, 0, stream>>>(zbuf, c0, out_h, out_c, Ahb, A2);

  // ---- 4. Vh = h_new @ V_w^T + V_b ----
  wmma_gemm_bf16<<<dim3((2 * Hn) / 64, Bn / 16), 32, 0, stream>>>(
      Ahb, Hn, Vwb, Hn, V_b, nullptr, Vh, 2 * Hn, Hn);

  // ---- 5. scores e[b,l] (HBM pass 1 over h_enc) ----
  attn_scores<<<dim3(Ln, Bn), 32, 0, stream>>>(Vh, henc, ebuf);

  // ---- 6. softmax over L ----
  softmax_rows<<<Bn, 256, 0, stream>>>(ebuf, albuf);

  // ---- 7. context vector (HBM pass 2 over h_enc, reinterpreted view) ----
  attn_context<<<Bn, 256, 0, stream>>>(albuf, henc, cvec, A2);

  // ---- 8. out = [c_vec | h_new] @ fc_w^T + fc_b ----
  wmma_gemm_bf16<<<dim3(OUTn / 64, Bn / 16), 32, 0, stream>>>(
      A2, 3 * Hn, fcwb, 3 * Hn, fc_b, nullptr, out_main, OUTn, 3 * Hn);
}
